// Proposal_Layer_vote_7473243095570
// MI455X (gfx1250) — matrix operand order repaired and checked by reading, not compiled, not run on hardware
//
#include <hip/hip_runtime.h>
#include <hip/hip_bf16.h>

// P2B proposal layer for MI455X (gfx1250, wave32, WMMA).
// Hot path: fused SharedMLP (layer1 collapsed algebraically; layers 2&3 as
// chained v_wmma_f32_16x16x32_f16 GEMMs with LDS-staged activations, 2x2x4
// register blocking, and fused max-over-template via int-atomicMax).
// Cosine x-corr uses WMMA on pre-transposed f16 operands (pure b128 loads).

typedef __attribute__((ext_vector_type(16))) _Float16 v16h;
typedef __attribute__((ext_vector_type(8)))  _Float16 v8h;
typedef __attribute__((ext_vector_type(8)))  float    v8f;

#define BB   2
#define N1   256
#define N2   1024
#define CH   256
#define NPROP 64
#define NSAMP 16

// ---------------------------------------------------------------------------
// prep: fold BN scale into f16 weight copies of mlp[1], mlp[2]
__global__ void prep_w_kernel(const float* __restrict__ W2, const float* __restrict__ s2,
                              const float* __restrict__ W3, const float* __restrict__ s3,
                              _Float16* __restrict__ W2h, _Float16* __restrict__ W3h) {
    int idx = blockIdx.x * 256 + threadIdx.x;
    if (idx >= CH * CH) return;
    int o = idx >> 8;
    W2h[idx] = (_Float16)(s2[o] * W2[idx]);
    W3h[idx] = (_Float16)(s3[o] * W3[idx]);
}

// prep: transposed f16 copies of features: [b][point][feature]
__global__ void prep_tr_kernel(const float* __restrict__ tfeat, const float* __restrict__ sfeat,
                               _Float16* __restrict__ tfeatT, _Float16* __restrict__ sfeatT) {
    int idx = blockIdx.x * 256 + threadIdx.x;
    if (idx < BB * N1 * CH) {
        int f = idx % CH, n = (idx / CH) % N1, b = idx / (CH * N1);
        tfeatT[idx] = (_Float16)tfeat[((size_t)b * CH + f) * N1 + n];
    } else if (idx < BB * N1 * CH + BB * N2 * CH) {
        int j = idx - BB * N1 * CH;
        int f = j % CH, n = (j / CH) % N2, b = j / (CH * N2);
        sfeatT[j] = (_Float16)sfeat[((size_t)b * CH + f) * N2 + n];
    }
}

// prep: layer-1 collapse.  a[o] = s1[o]*W1[o][0]
// cT[b][n1][o] = s1[o]*(W1[o][1:4].txyz + W1[o][4:260].tfeat[:,n1]) + b1[o]
__global__ __launch_bounds__(256) void prep_ac_kernel(
    const float* __restrict__ W1, const float* __restrict__ s1, const float* __restrict__ b1,
    const float* __restrict__ txyz, const float* __restrict__ tfeat,
    float* __restrict__ aVec, float* __restrict__ cT) {
    __shared__ float col[CH];
    int bn = blockIdx.x;          // b*N1 + n1
    int b  = bn / N1, n1 = bn % N1;
    int o  = threadIdx.x;
    col[o] = tfeat[((size_t)b * CH + o) * N1 + n1];
    __syncthreads();
    const float* wr = W1 + (size_t)o * 260;
    float acc = wr[1] * txyz[((size_t)b * N1 + n1) * 3 + 0]
              + wr[2] * txyz[((size_t)b * N1 + n1) * 3 + 1]
              + wr[3] * txyz[((size_t)b * N1 + n1) * 3 + 2];
    for (int f = 0; f < CH; ++f) acc += wr[4 + f] * col[f];
    cT[((size_t)b * N1 + n1) * CH + o] = s1[o] * acc + b1[o];
    if (bn == 0) aVec[o] = s1[o] * wr[0];
}

// reciprocal feature norms: 1 / max(||.||, EPS)
__global__ void norms_kernel(const float* __restrict__ tfeat, const float* __restrict__ sfeat,
                             float* __restrict__ itn, float* __restrict__ isn) {
    int idx = blockIdx.x * 256 + threadIdx.x;
    if (idx < BB * N1) {
        int b = idx / N1, n = idx % N1;
        float a = 0.f;
        for (int f = 0; f < CH; ++f) { float v = tfeat[((size_t)b * CH + f) * N1 + n]; a += v * v; }
        itn[idx] = 1.f / fmaxf(sqrtf(a), 1e-8f);
    } else if (idx < BB * N1 + BB * N2) {
        int j = idx - BB * N1;
        int b = j / N2, n = j % N2;
        float a = 0.f;
        for (int f = 0; f < CH; ++f) { float v = sfeat[((size_t)b * CH + f) * N2 + n]; a += v * v; }
        isn[j] = 1.f / fmaxf(sqrtf(a), 1e-8f);
    }
}

__global__ void zero_kernel(float* __restrict__ p, int n) {
    int i = blockIdx.x * 256 + threadIdx.x;
    if (i < n) p[i] = 0.f;
}

// ---------------------------------------------------------------------------
// cosine x-corr via WMMA: cos[b][n1][n2] = <t(:,n1),s(:,n2)> * itn * isn
// operands are pre-transposed f16 -> two b128 loads per operand per K-step.
__global__ __launch_bounds__(256) void xcorr_kernel(
    const _Float16* __restrict__ tfeatT, const _Float16* __restrict__ sfeatT,
    const float* __restrict__ itn, const float* __restrict__ isn,
    float* __restrict__ cosb) {
    int lane = threadIdx.x & 31;
    int wave = threadIdx.x >> 5;
    int tile = blockIdx.x * 8 + wave;          // BB*16*64 = 2048 tiles
    int b    = tile >> 10;
    int rem  = tile & 1023;
    int m1b  = (rem >> 6) << 4;                // n1 base
    int n2b  = (rem & 63) << 4;                // n2 base
    int colN = lane & 15;
    int ka0  = (lane < 16) ? 0 : 8;
    int kbB0 = (lane < 16) ? 0 : 16;
    const _Float16* arow = tfeatT + ((size_t)b * N1 + m1b + colN) * CH;
    const _Float16* brow = sfeatT + ((size_t)b * N2 + n2b + colN) * CH;
    v8f acc = {};
#pragma unroll
    for (int kk = 0; kk < 8; ++kk) {
        int kb = kk * 32;
        v16h Av, Bv;
        v8h alo = *(const v8h*)(arow + kb + ka0);
        v8h ahi = *(const v8h*)(arow + kb + ka0 + 16);
        v8h blo = *(const v8h*)(brow + kb + ka0);
        v8h bhi = *(const v8h*)(brow + kb + ka0 + 16);
#pragma unroll
        for (int i = 0; i < 8; ++i) {
            Av[i] = alo[i]; Av[8 + i] = ahi[i];
            Bv[i] = blo[i]; Bv[8 + i] = bhi[i];
        }
        acc = __builtin_amdgcn_wmma_f32_16x16x32_f16(false, Av, false, Bv,
                                                     (short)0, acc, false, false);
    }
#pragma unroll
    for (int r = 0; r < 8; ++r) {
        int mm = m1b + ((lane < 16) ? r : 8 + r);
        int n  = n2b + colN;
        cosb[((size_t)b * N1 + mm) * N2 + n] = acc[r] * itn[b * N1 + mm] * isn[b * N2 + n];
    }
}

// ---------------------------------------------------------------------------
// Fused SharedMLP. Block = one batch x 32 n2 columns x half of the n1 range.
// Per iteration: 2 n1 values -> 64 GEMM columns share each weight A-tile.
// Each wave owns 2 output-channel tiles x 4 column tiles (8 accumulators).
// One LDS activation buffer is reused for layer-2 input and output.
// Result merged across the two n1-halves with int atomicMax (values >= 0).
__global__ __launch_bounds__(256) void fused_mlp_kernel(
    const float* __restrict__ cosb, const float* __restrict__ aVec,
    const float* __restrict__ cT,
    const _Float16* __restrict__ W2h, const float* __restrict__ b2,
    const _Float16* __restrict__ W3h, const float* __restrict__ b3,
    int* __restrict__ fpre_i) {
    __shared__ _Float16 cosL[N1][32];       // 16 KB
    __shared__ _Float16 act[2][32][CH];     // 32 KB, reused out1 -> out2
    int t    = threadIdx.x;
    int lane = t & 31, wave = t >> 5;
    int half = blockIdx.x & 1;              // n1 half
    int n2bk = (blockIdx.x >> 1) & 31;
    int b    = blockIdx.x >> 6;
    int n2b  = n2bk * 32;
    int n1lo = half * 128;
    int colN = lane & 15;
    int och0 = wave * 32, och1 = och0 + 16;
    int ka0  = (lane < 16) ? 0 : 8;
    int kbB0 = (lane < 16) ? 0 : 16;

    for (int c = 0; c < 32; ++c)
        cosL[t][c] = (_Float16)cosb[((size_t)b * N1 + t) * N2 + n2b + c];
    float aT = aVec[t];
    float mx[2][2][8];                      // [och tile][col tile][row]
#pragma unroll
    for (int oi = 0; oi < 2; ++oi)
#pragma unroll
        for (int ct = 0; ct < 2; ++ct)
#pragma unroll
            for (int r = 0; r < 8; ++r) mx[oi][ct][r] = 0.f;
    __syncthreads();

    for (int n1 = n1lo; n1 < n1lo + 128; n1 += 2) {
        // ---- layer 1 (collapsed rank-1): thread t = channel t
        float cv0 = cT[((size_t)b * N1 + n1) * CH + t];
        float cv1 = cT[((size_t)b * N1 + n1 + 1) * CH + t];
#pragma unroll
        for (int c = 0; c < 32; ++c) {
            float v0 = aT * (float)cosL[n1][c] + cv0;
            float v1 = aT * (float)cosL[n1 + 1][c] + cv1;
            act[0][c][t] = (_Float16)(v0 > 0.f ? v0 : 0.f);
            act[1][c][t] = (_Float16)(v1 > 0.f ? v1 : 0.f);
        }
        __syncthreads();

        // ---- layer 2: A-tiles loaded once per K-step, reused for 4 B-tiles
        v8f acc[2][4];
#pragma unroll
        for (int oi = 0; oi < 2; ++oi)
#pragma unroll
            for (int j = 0; j < 4; ++j) acc[oi][j] = (v8f){};
        for (int kk = 0; kk < 8; ++kk) {
            int kb = kk * 32;
            v16h A0, A1;
            {
                const v8h* a0lo = (const v8h*)(W2h + (size_t)(och0 + colN) * CH + kb + ka0);
                const v8h* a0hi = (const v8h*)(W2h + (size_t)(och0 + colN) * CH + kb + ka0 + 16);
                const v8h* a1lo = (const v8h*)(W2h + (size_t)(och1 + colN) * CH + kb + ka0);
                const v8h* a1hi = (const v8h*)(W2h + (size_t)(och1 + colN) * CH + kb + ka0 + 16);
#pragma unroll
                for (int i = 0; i < 8; ++i) {
                    A0[i] = (*a0lo)[i]; A0[8 + i] = (*a0hi)[i];
                    A1[i] = (*a1lo)[i]; A1[8 + i] = (*a1hi)[i];
                }
            }
#pragma unroll
            for (int j = 0; j < 4; ++j) {
                int n1s = j >> 1, ct = j & 1;
                const v8h* bp = (const v8h*)(&act[n1s][ct * 16 + colN][kb + ka0]);
                v16h Bv;
#pragma unroll
                for (int i = 0; i < 8; ++i) { Bv[i] = bp[0][i]; Bv[8 + i] = bp[2][i]; }
                acc[0][j] = __builtin_amdgcn_wmma_f32_16x16x32_f16(false, A0, false, Bv,
                                                                   (short)0, acc[0][j], false, false);
                acc[1][j] = __builtin_amdgcn_wmma_f32_16x16x32_f16(false, A1, false, Bv,
                                                                   (short)0, acc[1][j], false, false);
            }
        }
        __syncthreads();   // all layer-2 reads of act done -> safe to overwrite

#pragma unroll
        for (int j = 0; j < 4; ++j) {
            int n1s = j >> 1, ct = j & 1;
#pragma unroll
            for (int r = 0; r < 8; ++r) {
                int ro = (lane < 16) ? r : 8 + r;
                float v0 = acc[0][j][r] + b2[och0 + ro]; v0 = v0 > 0.f ? v0 : 0.f;
                float v1 = acc[1][j][r] + b2[och1 + ro]; v1 = v1 > 0.f ? v1 : 0.f;
                act[n1s][ct * 16 + colN][och0 + ro] = (_Float16)v0;
                act[n1s][ct * 16 + colN][och1 + ro] = (_Float16)v1;
            }
        }
        __syncthreads();

        // ---- layer 3 + running max over n1
        v8f acd[2][4];
#pragma unroll
        for (int oi = 0; oi < 2; ++oi)
#pragma unroll
            for (int j = 0; j < 4; ++j) acd[oi][j] = (v8f){};
        for (int kk = 0; kk < 8; ++kk) {
            int kb = kk * 32;
            v16h A0, A1;
            {
                const v8h* a0lo = (const v8h*)(W3h + (size_t)(och0 + colN) * CH + kb + ka0);
                const v8h* a0hi = (const v8h*)(W3h + (size_t)(och0 + colN) * CH + kb + ka0 + 16);
                const v8h* a1lo = (const v8h*)(W3h + (size_t)(och1 + colN) * CH + kb + ka0);
                const v8h* a1hi = (const v8h*)(W3h + (size_t)(och1 + colN) * CH + kb + ka0 + 16);
#pragma unroll
                for (int i = 0; i < 8; ++i) {
                    A0[i] = (*a0lo)[i]; A0[8 + i] = (*a0hi)[i];
                    A1[i] = (*a1lo)[i]; A1[8 + i] = (*a1hi)[i];
                }
            }
#pragma unroll
            for (int j = 0; j < 4; ++j) {
                int n1s = j >> 1, ct = j & 1;
                const v8h* bp = (const v8h*)(&act[n1s][ct * 16 + colN][kb + ka0]);
                v16h Bv;
#pragma unroll
                for (int i = 0; i < 8; ++i) { Bv[i] = bp[0][i]; Bv[8 + i] = bp[2][i]; }
                acd[0][j] = __builtin_amdgcn_wmma_f32_16x16x32_f16(false, A0, false, Bv,
                                                                   (short)0, acd[0][j], false, false);
                acd[1][j] = __builtin_amdgcn_wmma_f32_16x16x32_f16(false, A1, false, Bv,
                                                                   (short)0, acd[1][j], false, false);
            }
        }
#pragma unroll
        for (int j = 0; j < 4; ++j) {
            int ct = j & 1;
#pragma unroll
            for (int r = 0; r < 8; ++r) {
                int ro = (lane < 16) ? r : 8 + r;
                float v0 = acd[0][j][r] + b3[och0 + ro]; v0 = v0 > 0.f ? v0 : 0.f;
                float v1 = acd[1][j][r] + b3[och1 + ro]; v1 = v1 > 0.f ? v1 : 0.f;
                mx[0][ct][r] = fmaxf(mx[0][ct][r], v0);
                mx[1][ct][r] = fmaxf(mx[1][ct][r], v1);
            }
        }
        __syncthreads();   // protect act before next iteration's layer-1 write
    }
    // merge the two n1-halves (post-ReLU values >= 0 -> int order == float order)
#pragma unroll
    for (int ct = 0; ct < 2; ++ct)
#pragma unroll
        for (int r = 0; r < 8; ++r) {
            int ro = (lane < 16) ? r : 8 + r;
            int n2 = n2b + ct * 16 + colN;
            atomicMax(&fpre_i[((size_t)b * CH + och0 + ro) * N2 + n2], __float_as_int(mx[0][ct][r]));
            atomicMax(&fpre_i[((size_t)b * CH + och1 + ro) * N2 + n2], __float_as_int(mx[1][ct][r]));
        }
}

// ---------------------------------------------------------------------------
// generic 1x1 conv: out[b][o][n] = act(scale[o]*<W[o,:],in[b,:,n]> + bias[o])
__global__ void conv1d_kernel(const float* __restrict__ in, const float* __restrict__ W,
                              const float* __restrict__ scale, const float* __restrict__ bias,
                              float* __restrict__ out, int Bn, int Cin, int Cout, int N, int relu) {
    int idx = blockIdx.x * 256 + threadIdx.x;
    int total = Bn * Cout * N;
    if (idx >= total) return;
    int n = idx % N, o = (idx / N) % Cout, b = idx / (N * Cout);
    const float* ip = in + (size_t)b * Cin * N + n;
    const float* wp = W + (size_t)o * Cin;
    float acc = 0.f;
    for (int k = 0; k < Cin; ++k) acc += wp[k] * ip[(size_t)k * N];
    if (scale) acc *= scale[o];
    if (bias)  acc += bias[o];
    if (relu)  acc = fmaxf(acc, 0.f);
    out[idx] = acc;
}

__global__ void build_fxf_kernel(const float* __restrict__ sxyz, const float* __restrict__ fused,
                                 float* __restrict__ fxf) {
    int idx = blockIdx.x * 256 + threadIdx.x;
    if (idx >= BB * 259 * N2) return;
    int n = idx % N2, c = (idx / N2) % 259, b = idx / (N2 * 259);
    float v = (c < 3) ? sxyz[((size_t)b * N2 + n) * 3 + c]
                      : fused[((size_t)b * CH + (c - 3)) * N2 + n];
    fxf[idx] = v;
}

// vote = fxf + off; vote_xyz -> d_out seg0; vote[3:] -> vfeat channels 1..256
__global__ void vote_finalize_kernel(const float* __restrict__ fxf, const float* __restrict__ off,
                                     float* __restrict__ dout0, float* __restrict__ vfeat) {
    int idx = blockIdx.x * 256 + threadIdx.x;
    if (idx >= BB * 259 * N2) return;
    int n = idx % N2, c = (idx / N2) % 259, b = idx / (N2 * 259);
    float v = fxf[idx] + off[idx];
    if (c < 3) dout0[((size_t)b * N2 + n) * 3 + c] = v;
    else       vfeat[((size_t)b * 257 + (c - 2)) * N2 + n] = v;
}

__global__ void score_kernel(const float* __restrict__ cla, float* __restrict__ vfeat) {
    int idx = blockIdx.x * 256 + threadIdx.x;
    if (idx >= BB * N2) return;
    int b = idx / N2, n = idx % N2;
    vfeat[((size_t)b * 257 + 0) * N2 + n] = 1.f / (1.f + expf(-cla[idx]));
}

// farthest point sampling, one block (1024 thr) per batch, LDS argmax tree
__global__ __launch_bounds__(1024) void fps_kernel(const float* __restrict__ vxyz,
                                                   int* __restrict__ fps_idx) {
    __shared__ float px[N2], py[N2], pz[N2], dists[N2], rd[N2];
    __shared__ int ri[N2];
    int b = blockIdx.x, t = threadIdx.x;
    px[t] = vxyz[((size_t)b * N2 + t) * 3 + 0];
    py[t] = vxyz[((size_t)b * N2 + t) * 3 + 1];
    pz[t] = vxyz[((size_t)b * N2 + t) * 3 + 2];
    dists[t] = 1e10f;
    if (t == 0) fps_idx[b * NPROP] = 0;
    __syncthreads();
    int last = 0;
    for (int i = 1; i < NPROP; ++i) {
        float dx = px[t] - px[last], dy = py[t] - py[last], dz = pz[t] - pz[last];
        float d = fminf(dists[t], dx * dx + dy * dy + dz * dz);
        dists[t] = d; rd[t] = d; ri[t] = t;
        __syncthreads();
        for (int s = 512; s > 0; s >>= 1) {
            if (t < s && rd[t + s] > rd[t]) { rd[t] = rd[t + s]; ri[t] = ri[t + s]; }
            __syncthreads();
        }
        last = ri[0];
        if (t == 0) fps_idx[b * NPROP + i] = last;
        __syncthreads();
    }
}

__global__ void centers_kernel(const float* __restrict__ vxyz, const int* __restrict__ fps_idx,
                               float* __restrict__ dout2) {
    int idx = blockIdx.x * 256 + threadIdx.x;
    if (idx >= BB * NPROP * 3) return;
    int c = idx % 3, p = (idx / 3) % NPROP, b = idx / (3 * NPROP);
    int gi = fps_idx[b * NPROP + p];
    dout2[idx] = vxyz[((size_t)b * N2 + gi) * 3 + c];
}

__global__ void ballq_kernel(const float* __restrict__ vxyz, const float* __restrict__ centers,
                             int* __restrict__ grp) {
    int idx = blockIdx.x * 256 + threadIdx.x;
    if (idx >= BB * NPROP) return;
    int b = idx / NPROP;
    float cx = centers[idx * 3 + 0], cy = centers[idx * 3 + 1], cz = centers[idx * 3 + 2];
    int cnt = 0, first = 0;
    for (int n = 0; n < N2 && cnt < NSAMP; ++n) {
        float dx = vxyz[((size_t)b * N2 + n) * 3 + 0] - cx;
        float dy = vxyz[((size_t)b * N2 + n) * 3 + 1] - cy;
        float dz = vxyz[((size_t)b * N2 + n) * 3 + 2] - cz;
        if (dx * dx + dy * dy + dz * dz < 0.09f) {
            if (cnt == 0) first = n;
            grp[idx * NSAMP + cnt] = n;
            ++cnt;
        }
    }
    for (int j = cnt; j < NSAMP; ++j) grp[idx * NSAMP + j] = first;
}

// group + SA shared MLP (260->256->256->256) + max over nsample
__global__ __launch_bounds__(256) void group_sa_kernel(
    const float* __restrict__ vxyz, const float* __restrict__ centers,
    const float* __restrict__ vfeat, const int* __restrict__ grp,
    const float* __restrict__ W1, const float* __restrict__ s1, const float* __restrict__ b1,
    const float* __restrict__ W2, const float* __restrict__ s2, const float* __restrict__ b2,
    const float* __restrict__ W3, const float* __restrict__ s3, const float* __restrict__ b3,
    float* __restrict__ pfeat) {
    __shared__ float bufA[260 * NSAMP];
    __shared__ float bufB[CH * NSAMP];
    __shared__ float bufC[CH * NSAMP];
    int bp = blockIdx.x, b = bp / NPROP;
    int t = threadIdx.x;
    for (int i = t; i < 260 * NSAMP; i += 256) {
        int ch = i >> 4, j = i & 15;
        int gi = grp[bp * NSAMP + j];
        float v = (ch < 3) ? (vxyz[((size_t)b * N2 + gi) * 3 + ch] - centers[bp * 3 + ch])
                           : vfeat[((size_t)b * 257 + (ch - 3)) * N2 + gi];
        bufA[ch * NSAMP + j] = v;
    }
    __syncthreads();
    { const float* wp = W1 + (size_t)t * 260; float sc = s1[t], bi = b1[t];
      for (int j = 0; j < NSAMP; ++j) { float a = 0.f;
          for (int k = 0; k < 260; ++k) a += wp[k] * bufA[k * NSAMP + j];
          bufB[t * NSAMP + j] = fmaxf(sc * a + bi, 0.f); } }
    __syncthreads();
    { const float* wp = W2 + (size_t)t * CH; float sc = s2[t], bi = b2[t];
      for (int j = 0; j < NSAMP; ++j) { float a = 0.f;
          for (int k = 0; k < CH; ++k) a += wp[k] * bufB[k * NSAMP + j];
          bufC[t * NSAMP + j] = fmaxf(sc * a + bi, 0.f); } }
    __syncthreads();
    { const float* wp = W3 + (size_t)t * CH; float sc = s3[t], bi = b3[t];
      for (int j = 0; j < NSAMP; ++j) { float a = 0.f;
          for (int k = 0; k < CH; ++k) a += wp[k] * bufC[k * NSAMP + j];
          bufA[t * NSAMP + j] = fmaxf(sc * a + bi, 0.f); } }
    __syncthreads();
    float m = 0.f;
    for (int j = 0; j < NSAMP; ++j) m = fmaxf(m, bufA[t * NSAMP + j]);
    pfeat[((size_t)b * CH + t) * NPROP + (bp % NPROP)] = m;
}

__global__ void boxes_kernel(const float* __restrict__ po, const float* __restrict__ centers,
                             float* __restrict__ dout1) {
    int idx = blockIdx.x * 256 + threadIdx.x;
    if (idx >= BB * NPROP * 4) return;
    int c = idx % 4, p = (idx / 4) % NPROP, b = idx / (4 * NPROP);
    float v = po[((size_t)b * 4 + c) * NPROP + p];
    if (c < 3) v += centers[((size_t)b * NPROP + p) * 3 + c];
    dout1[idx] = v;
}

// ---------------------------------------------------------------------------
extern "C" void kernel_launch(void* const* d_in, const int* in_sizes, int n_in,
                              void* d_out, int out_size, void* d_ws, size_t ws_size,
                              hipStream_t stream) {
    const float* sxyz  = (const float*)d_in[0];
    const float* sfeat = (const float*)d_in[1];
    const float* txyz  = (const float*)d_in[2];
    const float* tfeat = (const float*)d_in[3];
    // params flattened with sorted dict keys (jax pytree order):
    const float* claW0 = (const float*)d_in[4];  const float* claS0 = (const float*)d_in[5];  const float* claB0 = (const float*)d_in[6];
    const float* claW1 = (const float*)d_in[7];  const float* claS1 = (const float*)d_in[8];  const float* claB1 = (const float*)d_in[9];
    const float* claoW = (const float*)d_in[10]; const float* claoB = (const float*)d_in[11];
    const float* fea1W = (const float*)d_in[12]; const float* fea1S = (const float*)d_in[13]; const float* fea1B = (const float*)d_in[14];
    const float* fea2W = (const float*)d_in[15]; const float* fea2B = (const float*)d_in[16];
    const float* mlpW0 = (const float*)d_in[17]; const float* mlpS0 = (const float*)d_in[18]; const float* mlpB0 = (const float*)d_in[19];
    const float* mlpW1 = (const float*)d_in[20]; const float* mlpS1 = (const float*)d_in[21]; const float* mlpB1 = (const float*)d_in[22];
    const float* mlpW2 = (const float*)d_in[23]; const float* mlpS2 = (const float*)d_in[24]; const float* mlpB2 = (const float*)d_in[25];
    const float* prW0  = (const float*)d_in[26]; const float* prS0  = (const float*)d_in[27]; const float* prB0  = (const float*)d_in[28];
    const float* prW1  = (const float*)d_in[29]; const float* prS1  = (const float*)d_in[30]; const float* prB1  = (const float*)d_in[31];
    const float* proW  = (const float*)d_in[32]; const float* proB  = (const float*)d_in[33];
    const float* saW0  = (const float*)d_in[34]; const float* saS0  = (const float*)d_in[35]; const float* saB0  = (const float*)d_in[36];
    const float* saW1  = (const float*)d_in[37]; const float* saS1  = (const float*)d_in[38]; const float* saB1  = (const float*)d_in[39];
    const float* saW2  = (const float*)d_in[40]; const float* saS2  = (const float*)d_in[41]; const float* saB2  = (const float*)d_in[42];
    const float* vtW0  = (const float*)d_in[43]; const float* vtS0  = (const float*)d_in[44]; const float* vtB0  = (const float*)d_in[45];
    const float* vtW1  = (const float*)d_in[46]; const float* vtS1  = (const float*)d_in[47]; const float* vtB1  = (const float*)d_in[48];
    const float* vtoW  = (const float*)d_in[49]; const float* vtoB  = (const float*)d_in[50];

    // output segments (floats)
    float* dout0 = (float*)d_out;           // vote_xyz        (B,1024,3) 6144
    float* dout1 = dout0 + 6144;            // estimation_boxs (B,64,4)    512
    float* dout2 = dout1 + 512;             // center_xyzs     (B,64,3)    384
    float* dout3 = dout2 + 384;             // estimation_cla  (B,1,1024) 2048

    // workspace carve (256-byte aligned bump allocator)
    char* w = (char*)d_ws;
    size_t off = 0;
    auto alloc = [&](size_t bytes) { void* p = w + off; off += (bytes + 255) & ~(size_t)255; return p; };
    float*    cosb  = (float*)alloc((size_t)BB * N1 * N2 * 4);
    float*    aVec  = (float*)alloc(256 * 4);
    float*    cT    = (float*)alloc((size_t)BB * N1 * CH * 4);
    _Float16* W2h   = (_Float16*)alloc((size_t)CH * CH * 2);
    _Float16* W3h   = (_Float16*)alloc((size_t)CH * CH * 2);
    _Float16* tfT   = (_Float16*)alloc((size_t)BB * N1 * CH * 2);
    _Float16* sfT   = (_Float16*)alloc((size_t)BB * N2 * CH * 2);
    float*    itn   = (float*)alloc(BB * N1 * 4);
    float*    isn   = (float*)alloc(BB * N2 * 4);
    float*    fpre  = (float*)alloc((size_t)BB * CH * N2 * 4);
    float*    ftmp  = (float*)alloc((size_t)BB * CH * N2 * 4);
    float*    fused = (float*)alloc((size_t)BB * CH * N2 * 4);
    float*    t1    = (float*)alloc((size_t)BB * CH * N2 * 4);
    float*    t2    = (float*)alloc((size_t)BB * CH * N2 * 4);
    float*    fxf   = (float*)alloc((size_t)BB * 259 * N2 * 4);
    float*    vt1   = (float*)alloc((size_t)BB * CH * N2 * 4);
    float*    vt2   = (float*)alloc((size_t)BB * CH * N2 * 4);
    float*    voff  = (float*)alloc((size_t)BB * 259 * N2 * 4);
    float*    vfeat = (float*)alloc((size_t)BB * 257 * N2 * 4);
    int*      fpsi  = (int*)alloc(BB * NPROP * 4);
    int*      grp   = (int*)alloc(BB * NPROP * NSAMP * 4);
    float*    pfeat = (float*)alloc((size_t)BB * CH * NPROP * 4);
    float*    pt1   = (float*)alloc((size_t)BB * CH * NPROP * 4);
    float*    pt2   = (float*)alloc((size_t)BB * CH * NPROP * 4);
    float*    po    = (float*)alloc((size_t)BB * 4 * NPROP * 4);

    // 1) prep
    prep_w_kernel<<<(CH * CH + 255) / 256, 256, 0, stream>>>(mlpW1, mlpS1, mlpW2, mlpS2, W2h, W3h);
    prep_tr_kernel<<<(BB * CH * (N1 + N2) + 255) / 256, 256, 0, stream>>>(tfeat, sfeat, tfT, sfT);
    prep_ac_kernel<<<BB * N1, 256, 0, stream>>>(mlpW0, mlpS0, mlpB0, txyz, tfeat, aVec, cT);
    norms_kernel<<<(BB * (N1 + N2) + 255) / 256, 256, 0, stream>>>(tfeat, sfeat, itn, isn);
    zero_kernel<<<(BB * CH * N2 + 255) / 256, 256, 0, stream>>>(fpre, BB * CH * N2);
    // 2) xcorr (WMMA)
    xcorr_kernel<<<256, 256, 0, stream>>>(tfT, sfT, itn, isn, cosb);
    // 3) fused SharedMLP + max over template (WMMA hot loop)
    fused_mlp_kernel<<<BB * 32 * 2, 256, 0, stream>>>(cosb, aVec, cT, W2h, mlpB1, W3h, mlpB2,
                                                      (int*)fpre);
    // 4) fea layers
    conv1d_kernel<<<(BB * CH * N2 + 255) / 256, 256, 0, stream>>>(fpre, fea1W, fea1S, fea1B, ftmp, BB, CH, CH, N2, 1);
    conv1d_kernel<<<(BB * CH * N2 + 255) / 256, 256, 0, stream>>>(ftmp, fea2W, nullptr, fea2B, fused, BB, CH, CH, N2, 0);
    // 5) classification head -> d_out seg3 (logits)
    conv1d_kernel<<<(BB * CH * N2 + 255) / 256, 256, 0, stream>>>(fused, claW0, claS0, claB0, t1, BB, CH, CH, N2, 1);
    conv1d_kernel<<<(BB * CH * N2 + 255) / 256, 256, 0, stream>>>(t1, claW1, claS1, claB1, t2, BB, CH, CH, N2, 1);
    conv1d_kernel<<<(BB * 1 * N2 + 255) / 256, 256, 0, stream>>>(t2, claoW, nullptr, claoB, dout3, BB, CH, 1, N2, 0);
    // 6) voting
    build_fxf_kernel<<<(BB * 259 * N2 + 255) / 256, 256, 0, stream>>>(sxyz, fused, fxf);
    conv1d_kernel<<<(BB * CH * N2 + 255) / 256, 256, 0, stream>>>(fxf, vtW0, vtS0, vtB0, vt1, BB, 259, CH, N2, 1);
    conv1d_kernel<<<(BB * CH * N2 + 255) / 256, 256, 0, stream>>>(vt1, vtW1, vtS1, vtB1, vt2, BB, CH, CH, N2, 1);
    conv1d_kernel<<<(BB * 259 * N2 + 255) / 256, 256, 0, stream>>>(vt2, vtoW, nullptr, vtoB, voff, BB, CH, 259, N2, 0);
    vote_finalize_kernel<<<(BB * 259 * N2 + 255) / 256, 256, 0, stream>>>(fxf, voff, dout0, vfeat);
    score_kernel<<<(BB * N2 + 255) / 256, 256, 0, stream>>>(dout3, vfeat);
    // 7) FPS + centers + ball query
    fps_kernel<<<BB, 1024, 0, stream>>>(dout0, fpsi);
    centers_kernel<<<(BB * NPROP * 3 + 255) / 256, 256, 0, stream>>>(dout0, fpsi, dout2);
    ballq_kernel<<<(BB * NPROP + 255) / 256, 256, 0, stream>>>(dout0, dout2, grp);
    // 8) group + SA MLP + maxpool
    group_sa_kernel<<<BB * NPROP, 256, 0, stream>>>(dout0, dout2, vfeat, grp,
                                                    saW0, saS0, saB0, saW1, saS1, saB1,
                                                    saW2, saS2, saB2, pfeat);
    // 9) proposal head -> boxes
    conv1d_kernel<<<(BB * CH * NPROP + 255) / 256, 256, 0, stream>>>(pfeat, prW0, prS0, prB0, pt1, BB, CH, CH, NPROP, 1);
    conv1d_kernel<<<(BB * CH * NPROP + 255) / 256, 256, 0, stream>>>(pt1, prW1, prS1, prB1, pt2, BB, CH, CH, NPROP, 1);
    conv1d_kernel<<<(BB * 4 * NPROP + 255) / 256, 256, 0, stream>>>(pt2, proW, nullptr, proB, po, BB, CH, 4, NPROP, 0);
    boxes_kernel<<<(BB * NPROP * 4 + 255) / 256, 256, 0, stream>>>(po, dout2, dout1);
    (void)in_sizes; (void)n_in; (void)out_size; (void)ws_size;
}
